// LSTMTagger_41171556499836
// MI455X (gfx1250) — matrix operand order repaired and checked by reading, not compiled, not run on hardware
//
#include <hip/hip_runtime.h>
#include <hip/hip_bf16.h>

typedef __attribute__((ext_vector_type(16))) __bf16 v16bf;
typedef __attribute__((ext_vector_type(8)))  float  v8f;

#define T_TOK 8192
#define MM    12
#define HID   32
#define NTAGS 44

__device__ __forceinline__ float sigmoidf_(float x) { return 1.0f / (1.0f + __expf(-x)); }

// ---------------------------------------------------------------------------
// Pre-swizzle A (fp32 row-major [rows x lda], K zero-padded to KB*32) into
// WMMA fragment-native bf16 layout:
//   Aswz[ ((rowtile*KB + kb)*32 + lane)*16 + e ]
// where row = rowtile*16 + (lane&15),
//       k   = kb*32 + 16*(e>=8) + 8*(lane>=16) + (e&7)   (ISA 16-bit A layout)
// ---------------------------------------------------------------------------
__global__ void k_swz_a(const float* __restrict__ A, int lda, int Kreal, int KB,
                        __bf16* __restrict__ out, int total)
{
    for (int idx = blockIdx.x * blockDim.x + threadIdx.x; idx < total;
         idx += gridDim.x * blockDim.x) {
        int e    = idx & 15;
        int lane = (idx >> 4) & 31;
        int kb   = (idx >> 9) % KB;
        int i    = idx / (KB * 512);
        int row  = i * 16 + (lane & 15);
        int k    = kb * 32 + ((e >= 8) ? 16 : 0) + ((lane >= 16) ? 8 : 0) + (e & 7);
        float v  = (k < Kreal) ? A[(size_t)row * lda + k] : 0.0f;
        out[idx] = (__bf16)v;
    }
}

// ---------------------------------------------------------------------------
// Pre-swizzle weight W[2,128,Kreal] (fp32) into fragment-native bf16 B layout:
//   Bswz[ ((j*KB + kb)*32 + lane)*16 + e ]
// where n = j*16 + (lane&15)  (n = dir*128 + gate, 16 col tiles),
//       k = kb*32 + 16*(e>=8) + 8*(lane>=16) + (e&7)   (ISA 16-bit B layout)
// ---------------------------------------------------------------------------
__global__ void k_swz_b(const float* __restrict__ W, int Kreal, int KB,
                        __bf16* __restrict__ out)
{
    int idx = blockIdx.x * blockDim.x + threadIdx.x;
    int total = 16 * KB * 512;
    if (idx >= total) return;
    int e    = idx & 15;
    int lane = (idx >> 4) & 31;
    int kb   = (idx >> 9) % KB;
    int j    = idx / (KB * 512);
    int n    = j * 16 + (lane & 15);
    int k    = kb * 32 + ((e >= 8) ? 16 : 0) + ((lane >= 16) ? 8 : 0) + (e & 7);
    float v  = (k < Kreal) ? W[(size_t)n * Kreal + k] : 0.0f;
    out[idx] = (__bf16)v;
}

// ---------------------------------------------------------------------------
// C[rows x 256] = A @ W^T + bias, from pre-swizzled bf16 fragments.
// One wave per 16x32 output slab (two 16x16 col tiles sharing the A fragment).
// grid = (rows/16, 2), block = 128 (wave w owns col tiles by*8+w and by*8+w+4).
// KB is compile-time -> fully unrolled k-loop, 2*KB static wmma.
// ---------------------------------------------------------------------------
template <int KB>
__global__ void __launch_bounds__(128)
k_gemm_swz(const __bf16* __restrict__ Aswz, const __bf16* __restrict__ Bswz,
           const float* __restrict__ bias, float* __restrict__ C)
{
    const int lane = threadIdx.x & 31;
    const int wave = threadIdx.x >> 5;
    const int i    = blockIdx.x;
    const int j0   = blockIdx.y * 8 + wave;
    const int j1   = j0 + 4;
    const __bf16* ap  = Aswz + ((size_t)i  * KB * 32 + lane) * 16;
    const __bf16* bp0 = Bswz + ((size_t)j0 * KB * 32 + lane) * 16;
    const __bf16* bp1 = Bswz + ((size_t)j1 * KB * 32 + lane) * 16;

    v8f acc0 = {}, acc1 = {};
    #pragma unroll
    for (int kb = 0; kb < KB; ++kb) {
        v16bf a  = *(const v16bf*)(ap  + (size_t)kb * 512);
        v16bf b0 = *(const v16bf*)(bp0 + (size_t)kb * 512);
        v16bf b1 = *(const v16bf*)(bp1 + (size_t)kb * 512);
        acc0 = __builtin_amdgcn_wmma_f32_16x16x32_bf16(false, a, false, b0,
                                                       (short)0, acc0, false, false);
        acc1 = __builtin_amdgcn_wmma_f32_16x16x32_bf16(false, a, false, b1,
                                                       (short)0, acc1, false, false);
    }
    const int   row0 = i * 16 + ((lane >= 16) ? 8 : 0);
    const int   c0   = j0 * 16 + (lane & 15);
    const int   c1   = j1 * 16 + (lane & 15);
    const float bb0  = bias[c0];
    const float bb1  = bias[c1];
    #pragma unroll
    for (int v = 0; v < 8; ++v) {
        C[(size_t)(row0 + v) * 256 + c0] = acc0[v] + bb0;
        C[(size_t)(row0 + v) * 256 + c1] = acc1[v] + bb1;
    }
}

// ---------------------------------------------------------------------------
// Morpheme-stage recurrence: single workgroup, carry (h,c) threaded across
// all T tokens.  Threads 0-127 = forward dir, 128-255 = backward dir.
// ---------------------------------------------------------------------------
__global__ void __launch_bounds__(256)
k_morph_seq(const float* __restrict__ gm,      // [T*M, 256]
            const float* __restrict__ Whh0,    // [2,128,32]
            const float* __restrict__ Wih1,    // [2,128,64]
            const float* __restrict__ Whh1,    // [2,128,32]
            const float* __restrict__ b1,      // [256]
            const float* __restrict__ dp_emb, const int* __restrict__ dp_in,
            const float* __restrict__ feat,
            float* __restrict__ tokin)         // [T, 96]
{
    __shared__ float h0[2][HID], c0[2][HID], h1[2][HID], c1[2][HID];
    __shared__ float gates[2][128];
    __shared__ float out0[MM][64];
    __shared__ float tokv[64];
    const int tid = threadIdx.x;
    const int dir = tid >> 7;
    const int g   = tid & 127;
    if (tid < 64) {
        int d = tid >> 5, jj = tid & 31;
        h0[d][jj] = 0.f; c0[d][jj] = 0.f; h1[d][jj] = 0.f; c1[d][jj] = 0.f;
    }
    __syncthreads();

    for (int t = 0; t < T_TOK; ++t) {
        // ----- layer 0 (both directions concurrently) -----
        for (int s = 0; s < MM; ++s) {
            int m = dir ? (MM - 1 - s) : s;
            float acc = gm[((size_t)(t * MM + m)) * 256 + dir * 128 + g];
            const float* w = Whh0 + (dir * 128 + g) * HID;
            #pragma unroll
            for (int k = 0; k < HID; ++k) acc += w[k] * h0[dir][k];
            gates[dir][g] = acc;
            __syncthreads();
            if (g < HID) {
                float iv = sigmoidf_(gates[dir][g]);
                float fv = sigmoidf_(gates[dir][HID + g]);
                float gv = tanhf(gates[dir][2 * HID + g]);
                float ov = sigmoidf_(gates[dir][3 * HID + g]);
                float cn = fv * c0[dir][g] + iv * gv;
                float hn = ov * tanhf(cn);
                c0[dir][g] = cn; h0[dir][g] = hn;
                out0[m][dir * HID + g] = hn;
            }
            __syncthreads();
        }
        // ----- layer 1 -----
        for (int s = 0; s < MM; ++s) {
            int m = dir ? (MM - 1 - s) : s;
            float acc = b1[dir * 128 + g];
            const float* wi = Wih1 + (dir * 128 + g) * 64;
            #pragma unroll
            for (int k = 0; k < 64; ++k) acc += wi[k] * out0[m][k];
            const float* wh = Whh1 + (dir * 128 + g) * HID;
            #pragma unroll
            for (int k = 0; k < HID; ++k) acc += wh[k] * h1[dir][k];
            gates[dir][g] = acc;
            __syncthreads();
            if (g < HID) {
                float iv = sigmoidf_(gates[dir][g]);
                float fv = sigmoidf_(gates[dir][HID + g]);
                float gv = tanhf(gates[dir][2 * HID + g]);
                float ov = sigmoidf_(gates[dir][3 * HID + g]);
                float cn = fv * c1[dir][g] + iv * gv;
                float hn = ov * tanhf(cn);
                c1[dir][g] = cn; h1[dir][g] = hn;
                // out[-1] = [fwd output at m=11, bwd output after its FIRST step]
                if ((dir == 0 && s == MM - 1) || (dir == 1 && s == 0))
                    tokv[dir * HID + g] = hn;
            }
            __syncthreads();
        }
        // token-LSTM input: [tok_vec(64) | dp_emb(4) | feat(1) | zero pad -> 96]
        if (tid < 96) {
            float v;
            if (tid < 64)       v = tokv[tid];
            else if (tid < 68)  v = dp_emb[dp_in[t] * 4 + (tid - 64)];
            else if (tid == 68) v = feat[t];
            else                v = 0.0f;
            tokin[(size_t)t * 96 + tid] = v;
        }
        __syncthreads();
    }
}

// ---------------------------------------------------------------------------
// Token-stage recurrence: grid = 2 blocks (one per direction), block = 128.
// ---------------------------------------------------------------------------
__global__ void __launch_bounds__(128)
k_tok_seq(const float* __restrict__ gpre,   // [T,256] input-side preactivations
          const float* __restrict__ Whh,    // [2,128,32]
          float* __restrict__ out)          // [T,64]
{
    __shared__ float h[HID], c[HID], gates[128];
    const int dir = blockIdx.x;
    const int g   = threadIdx.x;
    if (g < HID) { h[g] = 0.f; c[g] = 0.f; }
    __syncthreads();
    const float* w = Whh + (dir * 128 + g) * HID;
    for (int s = 0; s < T_TOK; ++s) {
        int t = dir ? (T_TOK - 1 - s) : s;
        if (s + 1 < T_TOK) {
            int tn = dir ? (T_TOK - 2 - s) : (s + 1);
            __builtin_prefetch(gpre + (size_t)tn * 256 + dir * 128 + g, 0, 1);
        }
        float acc = gpre[(size_t)t * 256 + dir * 128 + g];
        #pragma unroll
        for (int k = 0; k < HID; ++k) acc += w[k] * h[k];
        gates[g] = acc;
        __syncthreads();
        if (g < HID) {
            float iv = sigmoidf_(gates[g]);
            float fv = sigmoidf_(gates[HID + g]);
            float gv = tanhf(gates[2 * HID + g]);
            float ov = sigmoidf_(gates[3 * HID + g]);
            float cn = fv * c[g] + iv * gv;
            float hn = ov * tanhf(cn);
            c[g] = cn; h[g] = hn;
            out[(size_t)t * 64 + dir * HID + g] = hn;
        }
        __syncthreads();
    }
}

// ---------------------------------------------------------------------------
// Head: logits, mask fix, relu, softmax.  One wave32 per token.
// ---------------------------------------------------------------------------
__global__ void __launch_bounds__(128)
k_head(const float* __restrict__ tok_out, const float* __restrict__ Wout,
       const float* __restrict__ bout, const int* __restrict__ mask,
       float* __restrict__ out)
{
    const int lane = threadIdx.x & 31;
    const int t    = blockIdx.x * 4 + (threadIdx.x >> 5);
    const float* x = tok_out + (size_t)t * 64;

    float l0 = bout[lane];
    #pragma unroll
    for (int k = 0; k < 64; ++k) l0 += x[k] * Wout[lane * 64 + k];

    bool  valid1 = lane < (NTAGS - 32);            // tags 32..43
    float l1 = -1e30f;
    if (valid1) {
        l1 = bout[32 + lane];
        #pragma unroll
        for (int k = 0; k < 64; ++k) l1 += x[k] * Wout[(32 + lane) * 64 + k];
        if (lane == NTAGS - 33 && mask[t] <= 0) l1 = 1.0f;  // force last logit
    }
    l0 = fmaxf(l0, 0.0f);
    if (valid1) l1 = fmaxf(l1, 0.0f);

    float mx = fmaxf(l0, valid1 ? l1 : -1e30f);
    #pragma unroll
    for (int o = 16; o > 0; o >>= 1) mx = fmaxf(mx, __shfl_xor(mx, o, 32));
    float e0 = __expf(l0 - mx);
    float e1 = valid1 ? __expf(l1 - mx) : 0.0f;
    float sm = e0 + e1;
    #pragma unroll
    for (int o = 16; o > 0; o >>= 1) sm += __shfl_xor(sm, o, 32);
    float inv = 1.0f / sm;
    out[(size_t)t * NTAGS + lane] = e0 * inv;
    if (valid1) out[(size_t)t * NTAGS + 32 + lane] = e1 * inv;
}

// ---------------------------------------------------------------------------
extern "C" void kernel_launch(void* const* d_in, const int* in_sizes, int n_in,
                              void* d_out, int out_size, void* d_ws, size_t ws_size,
                              hipStream_t stream)
{
    (void)in_sizes; (void)n_in; (void)out_size; (void)ws_size;
    const float* morp_embs = (const float*)d_in[0];
    const float* feat      = (const float*)d_in[1];
    const float* dp_emb    = (const float*)d_in[2];
    const float* mWih0     = (const float*)d_in[3];
    const float* mWhh0     = (const float*)d_in[4];
    const float* mb0       = (const float*)d_in[5];   // [2,128] == bias256
    const float* mWih1     = (const float*)d_in[6];
    const float* mWhh1     = (const float*)d_in[7];
    const float* mb1       = (const float*)d_in[8];
    const float* tWih0     = (const float*)d_in[9];
    const float* tWhh0     = (const float*)d_in[10];
    const float* tb0       = (const float*)d_in[11];
    const float* tWih1     = (const float*)d_in[12];
    const float* tWhh1     = (const float*)d_in[13];
    const float* tb1       = (const float*)d_in[14];
    const float* Wout      = (const float*)d_in[15];
    const float* bout      = (const float*)d_in[16];
    const int*   dp_in     = (const int*)d_in[17];
    const int*   mask      = (const int*)d_in[18];

    char*  ws  = (char*)d_ws;
    size_t off = 0;
    auto alloc = [&](size_t bytes) -> void* {
        void* p = ws + off;
        off += (bytes + 255) & ~(size_t)255;
        return p;
    };
    const int RT_M = (T_TOK * MM) / 16;   // 6144 row tiles (morph)
    const int RT_T = T_TOK / 16;          // 512 row tiles (token)

    float*  g_morp = (float*) alloc((size_t)T_TOK * MM * 256 * sizeof(float));
    __bf16* AswzM  = (__bf16*)alloc((size_t)RT_M * 4 * 512 * sizeof(__bf16));
    __bf16* BswzM0 = (__bf16*)alloc((size_t)16 * 4 * 512 * sizeof(__bf16));
    __bf16* BswzT0 = (__bf16*)alloc((size_t)16 * 3 * 512 * sizeof(__bf16));
    __bf16* BswzT1 = (__bf16*)alloc((size_t)16 * 2 * 512 * sizeof(__bf16));
    float*  tokin  = (float*) alloc((size_t)T_TOK * 96  * sizeof(float));
    __bf16* AswzT0 = (__bf16*)alloc((size_t)RT_T * 3 * 512 * sizeof(__bf16));
    float*  g_t0   = (float*) alloc((size_t)T_TOK * 256 * sizeof(float));
    float*  out0t  = (float*) alloc((size_t)T_TOK * 64  * sizeof(float));
    __bf16* AswzT1 = (__bf16*)alloc((size_t)RT_T * 2 * 512 * sizeof(__bf16));
    float*  g_t1   = (float*) alloc((size_t)T_TOK * 256 * sizeof(float));
    float*  tokout = (float*) alloc((size_t)T_TOK * 64  * sizeof(float));

    // --- swizzle weights into fragment-native bf16 B operands ---
    k_swz_b<<<dim3((16 * 4 * 512 + 255) / 256), 256, 0, stream>>>(mWih0, 100, 4, BswzM0);
    k_swz_b<<<dim3((16 * 3 * 512 + 255) / 256), 256, 0, stream>>>(tWih0,  69, 3, BswzT0);
    k_swz_b<<<dim3((16 * 2 * 512 + 255) / 256), 256, 0, stream>>>(tWih1,  64, 2, BswzT1);

    // --- morpheme layer-0 input projections (WMMA) ---
    {
        int total = RT_M * 4 * 512;
        k_swz_a<<<dim3((total + 255) / 256), 256, 0, stream>>>(
            morp_embs, 100, 100, 4, AswzM, total);
        k_gemm_swz<4><<<dim3(RT_M, 2), 128, 0, stream>>>(AswzM, BswzM0, mb0, g_morp);
    }

    // --- morpheme recurrence (sequential chain, carry across tokens) ---
    k_morph_seq<<<1, 256, 0, stream>>>(
        g_morp, mWhh0, mWih1, mWhh1, mb1, dp_emb, dp_in, feat, tokin);

    // --- token layer-0 input projections (WMMA) ---
    {
        int total = RT_T * 3 * 512;
        k_swz_a<<<dim3((total + 255) / 256), 256, 0, stream>>>(
            tokin, 96, 96, 3, AswzT0, total);
        k_gemm_swz<3><<<dim3(RT_T, 2), 128, 0, stream>>>(AswzT0, BswzT0, tb0, g_t0);
    }

    // --- token layer-0 recurrence (2 independent directions) ---
    k_tok_seq<<<2, 128, 0, stream>>>(g_t0, tWhh0, out0t);

    // --- token layer-1 input projections (WMMA) ---
    {
        int total = RT_T * 2 * 512;
        k_swz_a<<<dim3((total + 255) / 256), 256, 0, stream>>>(
            out0t, 64, 64, 2, AswzT1, total);
        k_gemm_swz<2><<<dim3(RT_T, 2), 128, 0, stream>>>(AswzT1, BswzT1, tb1, g_t1);
    }

    // --- token layer-1 recurrence ---
    k_tok_seq<<<2, 128, 0, stream>>>(g_t1, tWhh1, tokout);

    // --- head: logits + mask + relu + softmax ---
    k_head<<<dim3(T_TOK / 4), 128, 0, stream>>>(
        tokout, Wout, bout, mask, (float*)d_out);
}
